// QuantizedLinear_28166395528087
// MI455X (gfx1250) — matrix operand order repaired and checked by reading, not compiled, not run on hardware
//
#include <hip/hip_runtime.h>

// ---------------------------------------------------------------------------
// QuantizedLinear: out[t,o] = (sum_k x[t,k] * Wq[o,k]) * scale + bias[o]
//   x : 8192 x 4096 f32, Wq : 4096 x 4096 int32 (int8-valued), out f32.
// bf16 WMMA path: V_WMMA_F32_16X16X32_BF16 (wave32, D = A*B + C).
// B[k,n] = W[n,k]  => a B column is a contiguous W row: no transpose needed.
// Software pipeline: register-prefetch tile t+1 + double-buffered LDS, one
// barrier per K-step; global-load waits land after the WMMA block.
// ---------------------------------------------------------------------------

typedef __attribute__((ext_vector_type(16))) __bf16 v16bf;
typedef __attribute__((ext_vector_type(8)))  __bf16 v8bf;
typedef __attribute__((ext_vector_type(8)))  float  v8f;

#define M_DIM 8192
#define N_DIM 4096
#define K_DIM 4096

#define BM 128
#define BN 128
#define BK 32
#define KTILES (K_DIM / BK)     // 128
#define LDS_STRIDE 40           // BK + 8 bf16 pad: 16B-aligned rows, no bank conflicts

// ---- stage one 128x32 tile pair (x rows + W rows) from global into registers
__device__ __forceinline__ void stage_global(const float* __restrict__ X,
                                             const int*   __restrict__ Wq,
                                             int block_m, int block_n, int k0, int tid,
                                             float4 (&ra)[4], int4 (&rb)[4])
{
#pragma unroll
    for (int it = 0; it < 4; ++it) {
        const int idx = tid + it * 256;       // 0..1023 -> 128 rows x 8 vec4
        const int row = idx >> 3;
        const int c4  = (idx & 7) << 2;
        ra[it] = *reinterpret_cast<const float4*>(
            X + (size_t)(block_m + row) * K_DIM + k0 + c4);
        rb[it] = *reinterpret_cast<const int4*>(
            Wq + (size_t)(block_n + row) * K_DIM + k0 + c4);
        // speculative prefetch one more tile ahead (global_prefetch_b8; no counters)
        __builtin_prefetch(X  + (size_t)(block_m + row) * K_DIM + k0 + BK + c4, 0, 1);
        __builtin_prefetch(Wq + (size_t)(block_n + row) * K_DIM + k0 + BK + c4, 0, 1);
    }
}

// ---- convert staged registers to bf16 and store into an LDS buffer
__device__ __forceinline__ void store_lds(__bf16* __restrict__ sA, __bf16* __restrict__ sB,
                                          int tid, const float4 (&ra)[4], const int4 (&rb)[4])
{
#pragma unroll
    for (int it = 0; it < 4; ++it) {
        const int idx = tid + it * 256;
        const int row = idx >> 3;
        const int c4  = (idx & 7) << 2;
        __bf16* da = sA + row * LDS_STRIDE + c4;
        da[0] = (__bf16)ra[it].x; da[1] = (__bf16)ra[it].y;
        da[2] = (__bf16)ra[it].z; da[3] = (__bf16)ra[it].w;
        __bf16* db = sB + row * LDS_STRIDE + c4;
        db[0] = (__bf16)(float)rb[it].x; db[1] = (__bf16)(float)rb[it].y;
        db[2] = (__bf16)(float)rb[it].z; db[3] = (__bf16)(float)rb[it].w;
    }
}

// ---- build fragments (ISA 16-bit WMMA layouts) and run 8 WMMAs for one K-step
__device__ __forceinline__ void compute_tile(const __bf16* __restrict__ sA,
                                             const __bf16* __restrict__ sB,
                                             int wm, int wn, int l15, int khalf,
                                             v8f (&acc)[2][4])
{
    // A (16x32 MxK): lanes 0-15 row=l15 hold K {0..7,16..23}; lanes 16-31 the
    // same rows with K {8..15,24..31}.
    v16bf afrag[2];
#pragma unroll
    for (int mi = 0; mi < 2; ++mi) {
        const __bf16* p = sA + (wm * 32 + mi * 16 + l15) * LDS_STRIDE + khalf * 8;
        const v8bf lo = *reinterpret_cast<const v8bf*>(p);
        const v8bf hi = *reinterpret_cast<const v8bf*>(p + 16);
#pragma unroll
        for (int e = 0; e < 8; ++e) { afrag[mi][e] = lo[e]; afrag[mi][8 + e] = hi[e]; }
    }
    // B (32x16 KxN): lane = column N; lanes 0-15 hold K {0..7,16..23}, lanes 16-31 K {8..15,24..31}.
    // Column n of B == row n of the W tile (contiguous in sB).
    v16bf bfrag[4];
#pragma unroll
    for (int ni = 0; ni < 4; ++ni) {
        const __bf16* p = sB + (wn * 64 + ni * 16 + l15) * LDS_STRIDE + khalf * 8;
        const v8bf lo = *reinterpret_cast<const v8bf*>(p);
        const v8bf hi = *reinterpret_cast<const v8bf*>(p + 16);
#pragma unroll
        for (int e = 0; e < 8; ++e) { bfrag[ni][e] = lo[e]; bfrag[ni][8 + e] = hi[e]; }
    }
#pragma unroll
    for (int mi = 0; mi < 2; ++mi)
#pragma unroll
        for (int ni = 0; ni < 4; ++ni)
            acc[mi][ni] = __builtin_amdgcn_wmma_f32_16x16x32_bf16(
                false, afrag[mi], false, bfrag[ni],
                (short)0, acc[mi][ni], false, false);
}

__global__ __launch_bounds__(256)
void qlinear_wmma_bf16(const float* __restrict__ X,
                       const int*   __restrict__ Wq,
                       const float* __restrict__ wscale,
                       const float* __restrict__ bias,
                       float*       __restrict__ Out)
{
    __shared__ __align__(16) __bf16 sA0[BM * LDS_STRIDE];
    __shared__ __align__(16) __bf16 sB0[BN * LDS_STRIDE];
    __shared__ __align__(16) __bf16 sA1[BM * LDS_STRIDE];
    __shared__ __align__(16) __bf16 sB1[BN * LDS_STRIDE];

    const int tid  = threadIdx.x;
    const int lane = tid & 31;
    const int wave = tid >> 5;
    const int wm   = wave & 3;        // wave's 32-row strip within BM
    const int wn   = wave >> 2;       // wave's 64-col strip within BN

    const int block_n = blockIdx.x * BN;   // N fast-varying -> x stripe reuse in L2
    const int block_m = blockIdx.y * BM;

    const int khalf = lane >> 4;
    const int l15   = lane & 15;

    v8f acc[2][4];
#pragma unroll
    for (int i = 0; i < 2; ++i)
#pragma unroll
        for (int j = 0; j < 4; ++j)
#pragma unroll
            for (int e = 0; e < 8; ++e) acc[i][j][e] = 0.0f;

    float4 ra[4];
    int4   rb[4];

    // Prologue: tile 0 -> buf0
    stage_global(X, Wq, block_m, block_n, 0, tid, ra, rb);
    store_lds(sA0, sB0, tid, ra, rb);
    __syncthreads();

    // Steady state: process tiles (t, t+1) per iteration; KTILES = 128 is even.
    for (int t = 0; t < KTILES; t += 2) {
        // ---- stage A: compute tile t from buf0; fill buf1 with tile t+1 ----
        stage_global(X, Wq, block_m, block_n, (t + 1) * BK, tid, ra, rb);  // issue loads
        compute_tile(sA0, sB0, wm, wn, l15, khalf, acc);                   // WMMAs hide them
        store_lds(sA1, sB1, tid, ra, rb);                                  // loadcnt wait here
        __syncthreads();

        // ---- stage B: compute tile t+1 from buf1; fill buf0 with tile t+2 ----
        const bool more = (t + 2) < KTILES;
        if (more) stage_global(X, Wq, block_m, block_n, (t + 2) * BK, tid, ra, rb);
        compute_tile(sA1, sB1, wm, wn, l15, khalf, acc);
        if (more) store_lds(sA0, sB0, tid, ra, rb);
        __syncthreads();
    }

    // ---- epilogue: C/D layout = VGPR r: lanes 0-15 M=r, lanes 16-31 M=8+r ----
    const float scale = wscale[0];
    const int   rsel  = khalf << 3;           // 0 or 8
#pragma unroll
    for (int mi = 0; mi < 2; ++mi) {
        const int row0 = block_m + wm * 32 + mi * 16 + rsel;
#pragma unroll
        for (int ni = 0; ni < 4; ++ni) {
            const int col = block_n + wn * 64 + ni * 16 + l15;
            const float bv = bias[col];
#pragma unroll
            for (int r = 0; r < 8; ++r) {
                Out[(size_t)(row0 + r) * N_DIM + col] = acc[mi][ni][r] * scale + bv;
            }
        }
    }
}

extern "C" void kernel_launch(void* const* d_in, const int* in_sizes, int n_in,
                              void* d_out, int out_size, void* d_ws, size_t ws_size,
                              hipStream_t stream) {
    const float* X      = (const float*)d_in[0];
    const int*   Wq     = (const int*)  d_in[1];
    const float* wscale = (const float*)d_in[2];
    const float* bias   = (const float*)d_in[3];
    float*       Out    = (float*)d_out;

    dim3 grid(N_DIM / BN, M_DIM / BM);   // (32, 64), N fast-varying
    dim3 block(256);                     // 8 wave32s
    qlinear_wmma_bf16<<<grid, block, 0, stream>>>(X, Wq, wscale, bias, Out);
}
